// RoIHeads_80058190397670
// MI455X (gfx1250) — compile-verified, matrix operand order checked by reading, not run on hardware
//
#include <hip/hip_runtime.h>

typedef unsigned long long u64;
typedef __attribute__((ext_vector_type(16))) _Float16 v16h;
typedef __attribute__((ext_vector_type(8)))  float    v8f;

#define NB        2
#define NPROP     50000
#define NCLS      81
#define KSEL      2048
#define NDETS     100
#define SEGS      16
#define CAPSEG    32768
#define SCORE_TH  0.05f
#define MIN_SZ    0.01f
#define IMW       1333.0f
#define IMH       800.0f
#define BBOX_CLIPV 4.135166556742356f
#define LBL_OFF   1401.0f

// ---------------- workspace layout (bytes) ----------------
#define OFF_COUNTS 0u            // NB*SEGS u32
#define OFF_CUT    128u          // NB u32
#define OFF_SCORE  256u          // NB*KSEL f32   (16384 B)
#define OFF_LABEL  16640u        // NB*KSEL i32   (16384 B)
#define OFF_AREA   33024u        // NB*KSEL f32   (16384 B)
#define OFF_BOX    49408u        // NB*KSEL float4 (65536 B)
#define OFF_OBOX   114944u       // NB*KSEL float4 (65536 B)
#define OFF_MASK   180480u       // NB*KSEL*128 u16 (1 MB)
#define OFF_CAND   1229056u      // NB*SEGS*CAPSEG u64 (8 MB)

// torchvision BoxCoder.decode + clip to image, one candidate
__device__ __forceinline__ void decode_clip(const float4 pr, const float4 rg,
                                            float& x1, float& y1, float& x2, float& y2) {
  const float w  = pr.z - pr.x, h  = pr.w - pr.y;
  const float cx = pr.x + 0.5f * w, cy = pr.y + 0.5f * h;
  const float dx = rg.x * 0.1f, dy = rg.y * 0.1f;              // /WX, /WY
  const float dw = fminf(rg.z * 0.2f, BBOX_CLIPV);             // /WW, clamp
  const float dh = fminf(rg.w * 0.2f, BBOX_CLIPV);
  const float pcx = dx * w + cx, pcy = dy * h + cy;
  const float pw = __expf(dw) * w, ph = __expf(dh) * h;
  x1 = fminf(fmaxf(pcx - 0.5f * pw, 0.0f), IMW);
  y1 = fminf(fmaxf(pcy - 0.5f * ph, 0.0f), IMH);
  x2 = fminf(fmaxf(pcx + 0.5f * pw, 0.0f), IMW);
  y2 = fminf(fmaxf(pcy + 0.5f * ph, 0.0f), IMH);
}

// k0: zero the per-image/segment candidate counters + cutoffs
__global__ void k0_init(unsigned* __restrict__ p) {
  if (threadIdx.x < 64) p[threadIdx.x] = 0u;
}

// k1: one wave32 per proposal row: softmax over 81 logits, score-gated gather of
// regression, decode+clip+size filter, wave-aggregated compaction into segmented lists.
__global__ __launch_bounds__(256) void k1_score(const float* __restrict__ logits,
                                                const float* __restrict__ reg,
                                                const float* __restrict__ props,
                                                u64* __restrict__ cand,
                                                unsigned* __restrict__ counts) {
  __shared__ unsigned lcnt[2];
  __shared__ unsigned lbase[2];
  __shared__ unsigned wbase[8];
  const int tid  = threadIdx.x;
  const int lane = tid & 31;
  const int wv   = tid >> 5;
  if (tid < 2) lcnt[tid] = 0u;
  __syncthreads();

  const int row = blockIdx.x * 8 + wv;                    // [0, NB*NPROP)
  const int b   = (row >= NPROP) ? 1 : 0;
  const int n   = row - b * NPROP;

  const float* lrow = logits + (size_t)row * NCLS;
  const float l0 = lrow[lane];
  const float l1 = lrow[lane + 32];
  const float l2 = (lane < 17) ? lrow[lane + 64] : -3.0e38f;
  float mx = fmaxf(l0, fmaxf(l1, l2));
#pragma unroll
  for (int o = 16; o > 0; o >>= 1) mx = fmaxf(mx, __shfl_xor(mx, o));
  const float e0 = __expf(l0 - mx);
  const float e1 = __expf(l1 - mx);
  const float e2 = (lane < 17) ? __expf(l2 - mx) : 0.0f;
  float sm = e0 + e1 + e2;
#pragma unroll
  for (int o = 16; o > 0; o >>= 1) sm += __shfl_xor(sm, o);
  const float inv = 1.0f / sm;

  const float4 pr = ((const float4*)props)[row];
  u64  ent[3];
  bool v[3];
#pragma unroll
  for (int s3 = 0; s3 < 3; ++s3) {
    const int   c = lane + 32 * s3;
    const float e = (s3 == 0) ? e0 : ((s3 == 1) ? e1 : e2);
    const float p = e * inv;
    bool ok = (c >= 1) && (c <= 80) && (p > SCORE_TH);
    u64 en = 0ull;
    if (ok) {  // gather 16B of regression only for passing candidates
      const float4 rg = *(const float4*)(reg + (size_t)row * (NCLS * 4) + c * 4);
      float x1, y1, x2, y2;
      decode_clip(pr, rg, x1, y1, x2, y2);
      ok = ((x2 - x1) >= MIN_SZ) && ((y2 - y1) >= MIN_SZ);
      if (ok) {
        const unsigned idx = (unsigned)n * 80u + (unsigned)(c - 1);
        en = ((u64)__float_as_uint(p) << 32) | (u64)(~idx);  // desc score, asc index
      }
    }
    v[s3] = ok; ent[s3] = en;
  }

  unsigned off[3]; unsigned tot = 0u;
#pragma unroll
  for (int s3 = 0; s3 < 3; ++s3) {
    const u64 bal = __ballot(v[s3]);
    off[s3] = tot + (unsigned)__popcll(bal & ((1ull << lane) - 1ull));
    tot += (unsigned)__popcll(bal);
  }
  if (lane == 0) wbase[wv] = (tot != 0u) ? atomicAdd(&lcnt[b], tot) : 0u;
  __syncthreads();
  if (tid < 2) {
    const unsigned cc = lcnt[tid];
    lbase[tid] = (cc != 0u) ? atomicAdd(&counts[tid * SEGS + (blockIdx.x & (SEGS - 1))], cc) : 0u;
  }
  __syncthreads();
  const unsigned base = lbase[b] + wbase[wv];
  u64* cbase = cand + (size_t)(b * SEGS + (blockIdx.x & (SEGS - 1))) * CAPSEG;
#pragma unroll
  for (int s3 = 0; s3 < 3; ++s3) {
    if (v[s3]) {
      const unsigned p2 = base + off[s3];
      if (p2 < CAPSEG) cbase[p2] = ent[s3];
    }
  }
}

// k2: per image, 8192-bucket LDS histogram over top-13 score bits -> cutoff bucket
// such that count(bucket > T) < KSEL <= count(bucket >= T).
__global__ __launch_bounds__(1024) void k2_hist(const u64* __restrict__ cand,
                                                const unsigned* __restrict__ counts,
                                                unsigned* __restrict__ cut) {
  __shared__ unsigned hist[8192];
  __shared__ unsigned csum[1024];
  const int b = blockIdx.x, tid = threadIdx.x;
#pragma unroll
  for (int i = 0; i < 8; ++i) hist[tid + 1024 * i] = 0u;
  __syncthreads();
  for (int s = 0; s < SEGS; ++s) {
    unsigned m = counts[b * SEGS + s]; if (m > CAPSEG) m = CAPSEG;
    const u64* cb = cand + (size_t)(b * SEGS + s) * CAPSEG;
    for (unsigned i = tid; i < m; i += 1024u)
      atomicAdd(&hist[(unsigned)(cb[i] >> 51) & 0x1FFFu], 1u);
  }
  __syncthreads();
  unsigned sum = 0u;
#pragma unroll
  for (int k = 0; k < 8; ++k) sum += hist[tid * 8 + k];
  csum[tid] = sum;
  __syncthreads();
  if (tid == 0) {
    unsigned acc = 0u, T = 0u; bool found = false;
    for (int q = 1023; q >= 0 && !found; --q) {
      const unsigned cs = csum[q];
      if (cs != 0u && acc + cs >= (unsigned)KSEL) {
        for (int k = 7; k >= 0; --k) {
          const unsigned hc = hist[q * 8 + k];
          if (hc != 0u && acc + hc >= (unsigned)KSEL) { T = (unsigned)(q * 8 + k); found = true; break; }
          acc += hc;
        }
      } else acc += cs;
    }
    cut[b] = found ? T : 0u;
  }
}

// k3: per image: compact candidates >= cutoff (<=4096) into LDS, bitonic-sort
// descending on (scoreBits, ~idx), decode the top-2048, emit offset boxes + areas.
__global__ __launch_bounds__(1024) void k3_select(const u64* __restrict__ cand,
                                                  const unsigned* __restrict__ counts,
                                                  const unsigned* __restrict__ cut,
                                                  const float* __restrict__ reg,
                                                  const float* __restrict__ props,
                                                  float* __restrict__ selScore,
                                                  int* __restrict__ selLabel,
                                                  float* __restrict__ selArea,
                                                  float4* __restrict__ selBox,
                                                  float4* __restrict__ selObox) {
  __shared__ u64 list[2 * KSEL];
  __shared__ unsigned cA, cE;
  const int b = blockIdx.x, tid = threadIdx.x;
  if (tid == 0) { cA = 0u; cE = 0u; }
  for (int i = tid; i < 2 * KSEL; i += 1024) list[i] = 0ull;
  __syncthreads();
  const unsigned T = cut[b];
  for (int s = 0; s < SEGS; ++s) {
    unsigned m = counts[b * SEGS + s]; if (m > CAPSEG) m = CAPSEG;
    const u64* cb = cand + (size_t)(b * SEGS + s) * CAPSEG;
    for (unsigned i = tid; i < m; i += 1024u) {
      const u64 e = cb[i];
      const unsigned bkt = (unsigned)(e >> 51) & 0x1FFFu;
      if (bkt > T)       { const unsigned p = atomicAdd(&cA, 1u); if (p < (unsigned)KSEL) list[p] = e; }
      else if (bkt == T) { const unsigned p = atomicAdd(&cE, 1u); if (p < (unsigned)KSEL) list[KSEL + p] = e; }
    }
  }
  __syncthreads();
  for (unsigned k = 2; k <= 2 * KSEL; k <<= 1) {
    for (unsigned j = k >> 1; j > 0; j >>= 1) {
      for (unsigned i = tid; i < 2 * KSEL; i += 1024u) {
        const unsigned l = i ^ j;
        if (l > i) {
          const u64 a = list[i], c2 = list[l];
          const bool dsc = ((i & k) == 0u);          // descending sort
          if (dsc ? (a < c2) : (a > c2)) { list[i] = c2; list[l] = a; }
        }
      }
      __syncthreads();
    }
  }
  for (int s2 = tid; s2 < KSEL; s2 += 1024) {
    const u64 e = list[s2];
    const int o = b * KSEL + s2;
    if (e == 0ull) {
      selScore[o] = -1.0f; selLabel[o] = -1; selArea[o] = 0.0f;
      selBox[o]  = make_float4(0.f, 0.f, 0.f, 0.f);
      selObox[o] = make_float4(0.f, 0.f, 0.f, 0.f);
    } else {
      const float sc = __uint_as_float((unsigned)(e >> 32));
      const unsigned idx = ~((unsigned)(e & 0xFFFFFFFFull));
      const unsigned n = idx / 80u;
      const int c = (int)(idx % 80u) + 1;
      const size_t row = (size_t)b * NPROP + n;
      const float4 pr = ((const float4*)props)[row];
      const float4 rg = *(const float4*)(reg + row * (NCLS * 4) + c * 4);
      float x1, y1, x2, y2;
      decode_clip(pr, rg, x1, y1, x2, y2);
      selScore[o] = sc; selLabel[o] = c;
      selArea[o]  = (x2 - x1) * (y2 - y1);
      selBox[o]   = make_float4(x1, y1, x2, y2);
      const float of = (float)c * LBL_OFF;             // batched-NMS class offset
      selObox[o]  = make_float4(x1 + of, y1 + of, x2 + of, y2 + of);
    }
  }
}

// k4: suppression bitmap. One wave per 16x16 tile of the 2048x2048 IoU matrix.
// The union term area_i + area_j is a rank-2 outer-sum computed on the matrix pipe:
// D = [a_i, 1] x [1; a_j] via v_wmma_f32_16x16x32_f16 (areas scaled by 1/64 for f16
// range, accumulated in f32, unscaled). VALU does only min/max intersection work.
// Division-free threshold: inter/(S - inter + eps) > 0.5  <=>  3*inter > S + eps
// (denominator strictly positive), so no v_div_* sequence on the critical path.
// Bits assembled with wave32 ballots into per-tile u16 words (no atomics).
__global__ __launch_bounds__(256) void k4_iou(const float4* __restrict__ selObox,
                                              const float* __restrict__ selArea,
                                              unsigned short* __restrict__ mask16) {
  const int b = blockIdx.y;
  const int tid = threadIdx.x;
  const int lane = tid & 31;
  const int half = lane >> 4;
  const int l16 = lane & 15;
  const int tile = blockIdx.x * 8 + (tid >> 5);   // 16384 tiles per image
  const int ti = tile >> 7, tj = tile & 127;
  const int i0 = ti * 16, j0 = tj * 16;
  const int base = b * KSEL;

  const float aI = selArea[base + i0 + l16] * 0.015625f;
  const float aJ = selArea[base + j0 + l16] * 0.015625f;
  v16h A = {};
  v16h Bm = {};
  A[0]  = (half == 0) ? (_Float16)aI   : (_Float16)0.0f;  // K=0: area_i
  A[1]  = (half == 0) ? (_Float16)1.0f : (_Float16)0.0f;  // K=1: 1
  Bm[0] = (half == 0) ? (_Float16)1.0f : (_Float16)0.0f;  // K=0: 1
  Bm[1] = (half == 0) ? (_Float16)aJ   : (_Float16)0.0f;  // K=1: area_j
  v8f Cz = {};
  const v8f D = __builtin_amdgcn_wmma_f32_16x16x32_f16(false, A, false, Bm,
                                                       (short)0, Cz, false, false);

  const float4 cb = selObox[base + j0 + l16];
#pragma unroll
  for (int r = 0; r < 8; ++r) {
    const float4 rb = selObox[base + i0 + r + 8 * half];
    const float ix1 = fmaxf(cb.x, rb.x), iy1 = fmaxf(cb.y, rb.y);
    const float ix2 = fminf(cb.z, rb.z), iy2 = fminf(cb.w, rb.w);
    const float iw = fmaxf(ix2 - ix1, 0.0f), ih = fmaxf(iy2 - iy1, 0.0f);
    const float inter = iw * ih;
    // iou > 0.5  <=>  3*inter > areaSum + eps   (denominator > 0 always)
    const bool over = (3.0f * inter) > (D[r] * 64.0f + 1e-9f);
    const unsigned bal = (unsigned)__ballot(over);
    if (lane == 0) {
      mask16[(size_t)(base + i0 + r) * 128 + tj]     = (unsigned short)(bal & 0xFFFFu);
      mask16[(size_t)(base + i0 + r + 8) * 128 + tj] = (unsigned short)(bal >> 16);
    }
  }
}

// k5: greedy NMS scan (wave32: lane w owns 64 "removed" bits) + top-100 emission.
__global__ __launch_bounds__(32) void k5_scan_out(const float* __restrict__ selScore,
                                                  const int* __restrict__ selLabel,
                                                  const float4* __restrict__ selBox,
                                                  const unsigned short* __restrict__ mask16,
                                                  float* __restrict__ out) {
  __shared__ unsigned keepf[KSEL];
  const int b = blockIdx.x;
  const int lane = threadIdx.x;
  u64 removed = 0ull;
  const u64* m64 = (const u64*)mask16 + (size_t)b * KSEL * 32;
  const float* sc = selScore + b * KSEL;
#pragma unroll 4
  for (int i = 0; i < KSEL; ++i) {
    const u64 w = m64[(size_t)i * 32 + lane];                 // row i, cols [lane*64, +64)
    unsigned s = (unsigned)((removed >> (i & 63)) & 1ull);
    s = (unsigned)__shfl((int)s, i >> 6);
    const bool kp = (sc[i] > 0.0f) && (s == 0u);
    if (lane == 0) keepf[i] = kp ? 1u : 0u;
    if (kp) removed |= w;
  }
  __syncthreads();
  if (lane == 0) {
    float* ob = out + b * (NDETS * 4);
    float* os = out + NB * NDETS * 4 + b * NDETS;
    float* ol = out + NB * NDETS * 4 + NB * NDETS + b * NDETS;
    int k = 0;
    for (int i = 0; i < KSEL && k < NDETS; ++i) {
      if (keepf[i]) {
        const float4 bx = selBox[b * KSEL + i];
        ob[k * 4 + 0] = bx.x; ob[k * 4 + 1] = bx.y; ob[k * 4 + 2] = bx.z; ob[k * 4 + 3] = bx.w;
        os[k] = sc[i];
        ol[k] = (float)selLabel[b * KSEL + i];
        ++k;
      }
    }
    for (; k < NDETS; ++k) {
      ob[k * 4 + 0] = 0.f; ob[k * 4 + 1] = 0.f; ob[k * 4 + 2] = 0.f; ob[k * 4 + 3] = 0.f;
      os[k] = 0.f; ol[k] = -1.0f;
    }
  }
}

extern "C" void kernel_launch(void* const* d_in, const int* in_sizes, int n_in,
                              void* d_out, int out_size, void* d_ws, size_t ws_size,
                              hipStream_t stream) {
  (void)in_sizes; (void)n_in; (void)out_size; (void)ws_size;
  const float* logits = (const float*)d_in[0];   // [B*N, 81]
  const float* reg    = (const float*)d_in[1];   // [B*N, 324]
  const float* props  = (const float*)d_in[2];   // [B, N, 4]
  char* ws = (char*)d_ws;
  unsigned*       counts   = (unsigned*)(ws + OFF_COUNTS);
  unsigned*       cut      = (unsigned*)(ws + OFF_CUT);
  float*          selScore = (float*)(ws + OFF_SCORE);
  int*            selLabel = (int*)(ws + OFF_LABEL);
  float*          selArea  = (float*)(ws + OFF_AREA);
  float4*         selBox   = (float4*)(ws + OFF_BOX);
  float4*         selObox  = (float4*)(ws + OFF_OBOX);
  unsigned short* mask16   = (unsigned short*)(ws + OFF_MASK);
  u64*            cand     = (u64*)(ws + OFF_CAND);

  k0_init<<<1, 64, 0, stream>>>((unsigned*)ws);
  k1_score<<<(NB * NPROP) / 8, 256, 0, stream>>>(logits, reg, props, cand, counts);
  k2_hist<<<NB, 1024, 0, stream>>>(cand, counts, cut);
  k3_select<<<NB, 1024, 0, stream>>>(cand, counts, cut, reg, props,
                                     selScore, selLabel, selArea, selBox, selObox);
  k4_iou<<<dim3(2048, NB), 256, 0, stream>>>(selObox, selArea, mask16);
  k5_scan_out<<<NB, 32, 0, stream>>>(selScore, selLabel, selBox, mask16, (float*)d_out);
}
